// EvoFill_69913477644865
// MI455X (gfx1250) — matrix operand order repaired and checked by reading, not compiled, not run on hardware
//
#include <hip/hip_runtime.h>

// ---------------------------------------------------------------------------
// Types for CDNA5 WMMA
// ---------------------------------------------------------------------------
typedef __attribute__((ext_vector_type(16))) _Float16 v16h;
typedef __attribute__((ext_vector_type(8)))  _Float16 v8h;
typedef __attribute__((ext_vector_type(8)))  float    v8f;

// ---------------------------------------------------------------------------
// Device math helpers
// ---------------------------------------------------------------------------
__device__ __forceinline__ float gelu_f(float x) {
    return 0.5f * x * (1.f + erff(x * 0.70710678118654752440f));
}
__device__ __forceinline__ float silu_f(float x) {
    return x / (1.f + __expf(-x));
}
__device__ __forceinline__ float softplus_f(float x) {
    return (x > 20.f) ? x : log1pf(__expf(x));
}
__device__ __forceinline__ float block_reduce_sum(float v, float* sbuf) {
    int tid = threadIdx.x;
    sbuf[tid] = v; __syncthreads();
    for (int s = blockDim.x >> 1; s > 0; s >>= 1) {
        if (tid < s) sbuf[tid] += sbuf[tid + s];
        __syncthreads();
    }
    float r = sbuf[0]; __syncthreads();
    return r;
}

// ---------------------------------------------------------------------------
// Conversions
// ---------------------------------------------------------------------------
__global__ void k_f32_to_f16(const float* __restrict__ src, _Float16* __restrict__ dst, long n) {
    long i = (long)blockIdx.x * 256 + threadIdx.x;
    if (i < n) dst[i] = (_Float16)src[i];
}
// (O, I, KT) f32 -> (KT, O, I) f16  (per-tap weight matrices for tap-GEMM conv)
__global__ void k_convw(const float* __restrict__ w, _Float16* __restrict__ dst, int O, int I, int KT) {
    long i = (long)blockIdx.x * 256 + threadIdx.x;
    long tot = (long)O * I * KT;
    if (i >= tot) return;
    int kt = (int)(i % KT); long r = i / KT;
    int ii = (int)(r % I);  int o = (int)(r / I);
    dst[((long)kt * O + o) * I + ii] = (_Float16)w[i];
}

// ---------------------------------------------------------------------------
// WMMA GEMM with optional time-tap accumulation (SAME-padded conv as GEMM):
//   C[m,n] = act( sum_kt sum_k A[shift(m,kt), k] * W[kt,n,k] + bias[n] ) + resid[m,n]
// A: (M,K) f16 row-major, rows are (b,t) with t = m % seqlen; tap kt shifts t by kt-pad
// W: (taps, N, K) f16. C: f32, row stride ldc. Block tile 128x64, wave tile 16x64.
// ---------------------------------------------------------------------------
#define GLDA 40   // halfs: 32 K + 8 pad (80B rows, 16B aligned)
#define GLDB 72   // halfs: 64 N + 8 pad (144B rows, 16B aligned)

__global__ __launch_bounds__(256)
void k_gemm_wmma(const _Float16* __restrict__ A, const _Float16* __restrict__ W,
                 const float* __restrict__ bias, const float* __restrict__ resid,
                 float* __restrict__ C,
                 int M, int Nn, int K, int ldc,
                 int seqlen, int taps, int pad, int act) {
    __shared__ _Float16 As[128 * GLDA];
    __shared__ _Float16 Bs[32 * GLDB];

    const int tid = threadIdx.x;
    const int blockM = blockIdx.y * 128;
    const int blockN = blockIdx.x * 64;
    const int wid  = tid >> 5;
    const int lane = tid & 31;

    v8f acc[4];
    for (int j = 0; j < 4; ++j)
        for (int v = 0; v < 8; ++v) acc[j][v] = 0.f;

    for (int kt = 0; kt < taps; ++kt) {
        const _Float16* Wt = W + (size_t)kt * Nn * K;
        for (int k0 = 0; k0 < K; k0 += 32) {
            // ---- stage A tile (128 x 32): thread -> (row, 16-half chunk) ----
            {
                int r  = tid >> 1;
                int ko = (tid & 1) * 16;
                int m  = blockM + r;
                bool valid = false; long arow = 0;
                if (m < M) {
                    int t  = m % seqlen;
                    int bb = m / seqlen;
                    int st = t + kt - pad;
                    if (st >= 0 && st < seqlen) { valid = true; arow = (long)bb * seqlen + st; }
                }
                if (valid) {
                    const _Float16* ap = &A[arow * K + k0 + ko];
                    __builtin_prefetch(ap + 32, 0, 0);
                    *(v8h*)&As[r * GLDA + ko]     = *(const v8h*)ap;
                    *(v8h*)&As[r * GLDA + ko + 8] = *(const v8h*)(ap + 8);
                } else {
                    for (int q = 0; q < 16; ++q) As[r * GLDA + ko + q] = (_Float16)0.f;
                }
            }
            // ---- stage B tile K-major (32 x 64): thread -> (n, 8-K chunk) ----
            {
                int n  = tid >> 2;
                int kq = (tid & 3) * 8;
                int gn = blockN + n;
                _Float16 tmp[8];
                if (gn < Nn) {
                    *(v8h*)tmp = *(const v8h*)&Wt[(size_t)gn * K + k0 + kq];
                } else {
                    for (int q = 0; q < 8; ++q) tmp[q] = (_Float16)0.f;
                }
                for (int q = 0; q < 8; ++q) Bs[(kq + q) * GLDB + n] = tmp[q];
            }
            __syncthreads();

            // ---- fragments + 4 WMMAs per wave ----
            {
                const int arow2 = wid * 16 + (lane & 15);
                const int half  = lane >> 4;
                union AU { v16h v; v8h h[2]; } af, bf;
                // A frag: lane holds row M=lane%16; K = {half*8..+8} U {16+half*8..+8}
                af.h[0] = *(const v8h*)&As[arow2 * GLDA + half * 8];
                af.h[1] = *(const v8h*)&As[arow2 * GLDA + half * 8 + 16];
                for (int j = 0; j < 4; ++j) {
                    // B frag: lane holds col N=lane%16; K = {half*8..+8} U {16+half*8..+8}
                    for (int e = 0; e < 16; ++e)
                        bf.v[e] = Bs[(half * 8 + (e & 7) + (e >> 3) * 16) * GLDB + j * 16 + (lane & 15)];
                    acc[j] = __builtin_amdgcn_wmma_f32_16x16x32_f16(
                        false, af.v, false, bf.v, (short)0, acc[j], false, false);
                }
            }
            __syncthreads();
        }
    }

    // ---- epilogue (D layout: lanes 0-15 M=v, lanes 16-31 M=v+8; N=lane%16) ----
    for (int j = 0; j < 4; ++j) {
        for (int v = 0; v < 8; ++v) {
            int m = blockM + wid * 16 + v + (lane >> 4) * 8;
            int n = blockN + j * 16 + (lane & 15);
            if (m < M && n < Nn) {
                float val = acc[j][v] + (bias ? bias[n] : 0.f);
                if (act == 1)      val = gelu_f(val);
                else if (act == 2) val = silu_f(val);
                if (resid) val += resid[(size_t)m * Nn + n];
                C[(size_t)m * ldc + n] = val;
            }
        }
    }
}

// ---------------------------------------------------------------------------
// LayerNorm over last dim (blockDim == d), optional pre-add + dual output
// ---------------------------------------------------------------------------
__global__ void k_layernorm(const float* __restrict__ x, const float* __restrict__ add,
                            const float* __restrict__ g, const float* __restrict__ b,
                            float* __restrict__ outf, _Float16* __restrict__ outh, int d) {
    __shared__ float sbuf[256];
    long r = blockIdx.x;
    int tid = threadIdx.x;
    float v = x[r * d + tid];
    if (add) v += add[r * d + tid];
    float mu  = block_reduce_sum(v, sbuf) / (float)d;
    float dv  = v - mu;
    float var = block_reduce_sum(dv * dv, sbuf) / (float)d;
    float o = dv * rsqrtf(var + 1e-5f) * g[tid] + b[tid];
    if (outf) outf[r * d + tid] = o;
    if (outh) outh[r * d + tid] = (_Float16)o;
}

// ---------------------------------------------------------------------------
// BatchNorm (training-mode): stats per channel over all (b,t); then apply
// ---------------------------------------------------------------------------
__global__ void k_bn_stats(const float* __restrict__ x, float* __restrict__ mean,
                           float* __restrict__ var, int M, int d) {
    __shared__ float s1[256], s2[256];
    int c = blockIdx.x, tid = threadIdx.x;
    float a = 0.f, q = 0.f;
    for (long m = tid; m < M; m += 256) { float v = x[m * d + c]; a += v; q += v * v; }
    s1[tid] = a; s2[tid] = q; __syncthreads();
    for (int s = 128; s > 0; s >>= 1) {
        if (tid < s) { s1[tid] += s1[tid + s]; s2[tid] += s2[tid + s]; }
        __syncthreads();
    }
    if (tid == 0) { float mu = s1[0] / M; mean[c] = mu; var[c] = s2[0] / M - mu * mu; }
}
__global__ void k_bn_apply(const float* __restrict__ x, const float* __restrict__ mean,
                           const float* __restrict__ var, const float* __restrict__ g,
                           const float* __restrict__ b, int post_gelu,
                           float* __restrict__ outf, _Float16* __restrict__ outh,
                           long total, int d) {
    long i = (long)blockIdx.x * 256 + threadIdx.x;
    if (i >= total) return;
    int c = (int)(i % d);
    float v = (x[i] - mean[c]) * rsqrtf(var[c] + 1e-5f) * g[c] + b[c];
    if (post_gelu) v = gelu_f(v);
    if (outf) outf[i] = v;
    if (outh) outh[i] = (_Float16)v;
}

// ---------------------------------------------------------------------------
// Mamba2 pieces
// ---------------------------------------------------------------------------
__global__ void k_dt(const float* __restrict__ zx, const float* __restrict__ dtb,
                     float* __restrict__ dt, long total, int d_in, int off, int H) {
    long i = (long)blockIdx.x * 256 + threadIdx.x;
    if (i >= total) return;
    int h = (int)(i % H); long m = i / H;
    dt[i] = softplus_f(zx[m * d_in + off + h] + dtb[h]);
}
// causal depthwise conv (Kc taps) + SiLU on xBC channels, routed to x / B / C
__global__ void k_dwconv(const float* __restrict__ zx, const float* __restrict__ cw,
                         const float* __restrict__ cb,
                         float* __restrict__ xo, float* __restrict__ Bo, float* __restrict__ Co,
                         long total, int seqlen, int d_in, int di, int N, int Kc) {
    long i = (long)blockIdx.x * 256 + threadIdx.x;
    if (i >= total) return;
    int convdim = di + 2 * N;
    int c = (int)(i % convdim); long m = i / convdim;
    int t = (int)(m % seqlen);  long b = m / seqlen;
    float acc = cb[c];
    for (int k = 0; k < Kc; ++k) {
        int tt = t - (Kc - 1) + k;
        if (tt >= 0) acc += zx[(b * seqlen + tt) * d_in + di + c] * cw[c * Kc + k];
    }
    float v = silu_f(acc);
    if (c < di)          xo[m * di + c] = v;
    else if (c < di + N) Bo[m * N + (c - di)] = v;
    else                 Co[m * N + (c - di - N)] = v;
}
// selective scan: grid = B*H blocks, block = P threads; state h[N] in registers
template <int NN>
__global__ void k_scan(const float* __restrict__ x, const float* __restrict__ dt,
                       const float* __restrict__ Bm, const float* __restrict__ Cm,
                       const float* __restrict__ A_log, const float* __restrict__ Dp,
                       float* __restrict__ y, int T, int H, int P) {
    __shared__ float sB[NN], sC[NN];
    __shared__ float sdt;
    int b = blockIdx.x / H, h = blockIdx.x % H;
    int p = threadIdx.x;
    float A  = -__expf(A_log[h]);
    float Dv = Dp[h];
    float hs[NN];
#pragma unroll
    for (int n = 0; n < NN; ++n) hs[n] = 0.f;
    const int di = H * P;
    for (int t = 0; t < T; ++t) {
        long base = (long)b * T + t;
        if (p < NN) { sB[p] = Bm[base * NN + p]; sC[p] = Cm[base * NN + p]; }
        if (p == 0) sdt = dt[base * H + h];
        __syncthreads();
        float xv  = x[base * di + h * P + p];
        float dtv = sdt;
        float dA  = __expf(dtv * A);
        float dbx = dtv * xv;
        float yv  = 0.f;
#pragma unroll
        for (int n = 0; n < NN; ++n) { hs[n] = hs[n] * dA + dbx * sB[n]; yv += hs[n] * sC[n]; }
        y[base * di + h * P + p] = yv + Dv * xv;
        __syncthreads();
    }
}
// gated RMSNorm: out16 = (y * silu(z)) * rsqrt(mean(.^2)+eps) * norm_w
__global__ void k_gate_rms(const float* __restrict__ y, const float* __restrict__ zx, int zstride,
                           const float* __restrict__ nw, _Float16* __restrict__ out, int di) {
    __shared__ float sbuf[256];
    __shared__ float gb[512];
    long r = blockIdx.x; int tid = threadIdx.x;
    float ss = 0.f;
    for (int c = tid; c < di; c += 256) {
        float g = y[r * di + c] * silu_f(zx[r * zstride + c]);
        gb[c] = g; ss += g * g;
    }
    float tot = block_reduce_sum(ss, sbuf);
    float sc = rsqrtf(tot / (float)di + 1e-5f);
    for (int c = tid; c < di; c += 256)
        out[r * di + c] = (_Float16)(gb[c] * sc * nw[c]);
}

// ---------------------------------------------------------------------------
// Small data movers
// ---------------------------------------------------------------------------
__global__ void k_outer(const float* __restrict__ xe, const float* __restrict__ w,
                        float* __restrict__ out, long total, int d) {
    long i = (long)blockIdx.x * 256 + threadIdx.x;
    if (i >= total) return;
    int c = (int)(i % d); long r = i / d;
    out[i] = xe[r] * w[c];
}
__global__ void k_flip(float* __restrict__ dst, int dpitch, const float* __restrict__ src,
                       int spitch, long total, int T, int d) {
    long i = (long)blockIdx.x * 256 + threadIdx.x;
    if (i >= total) return;
    int c = (int)(i % d); long r = i / d;
    int t = (int)(r % T); long b = r / T;
    dst[(b * T + (T - 1 - t)) * (long)dpitch + c] = src[(b * T + t) * (long)spitch + c];
}
__global__ void k_copy_seq(float* __restrict__ dst, int Tdst, int tOff,
                           const float* __restrict__ src, int Tsrc, int sOff,
                           const float* __restrict__ add, long total, int T, int d) {
    long i = (long)blockIdx.x * 256 + threadIdx.x;
    if (i >= total) return;
    int c = (int)(i % d); long r = i / d;
    int t = (int)(r % T); long b = r / T;
    float v = src[((long)b * Tsrc + sOff + t) * d + c];
    if (add) v += add[((long)b * T + t) * d + c];
    dst[((long)b * Tdst + tOff + t) * d + c] = v;
}
__global__ void k_cat2(float* __restrict__ out, const float* __restrict__ a,
                       const float* __restrict__ b, long rows, int d) {
    long i = (long)blockIdx.x * 256 + threadIdx.x;
    long tot = rows * 2 * d;
    if (i >= tot) return;
    int c = (int)(i % (2 * d)); long r = i / (2 * d);
    out[i] = (c < d) ? a[r * d + c] : b[r * d + (c - d)];
}

// ===========================================================================
// Host side
// ===========================================================================
namespace {

struct Bump {
    char* base; size_t off;
    void* take(size_t bytes) {
        size_t a = (off + 255) & ~(size_t)255;
        void* r = base + a;
        off = a + bytes;
        return r;
    }
};

inline int grid1(long n) { return (int)((n + 255) / 256); }

inline void cvt16(hipStream_t s, const float* src, _Float16* dst, long n) {
    k_f32_to_f16<<<grid1(n), 256, 0, s>>>(src, dst, n);
}

inline void gemm(hipStream_t s, const _Float16* A, const _Float16* W, const float* bias,
                 const float* resid, float* C, int M, int N, int K, int ldc,
                 int seqlen, int taps, int pad, int act) {
    dim3 g((N + 63) / 64, (M + 127) / 128);
    k_gemm_wmma<<<g, 256, 0, s>>>(A, W, bias, resid, C, M, N, K, ldc, seqlen, taps, pad, act);
}

// ----- input leaf indices (alphabetical pytree flatten of setup_inputs) -----
enum {
    IN_X = 0, IN_XE = 1,
    BM_FFN_B1 = 2, BM_FFN_B2, BM_FFN_W1, BM_FFN_W2,
    BM_LN1_G, BM_LN1_B, BM_LN2_G, BM_LN2_B,
    BM_BWD = 10,   // A_log, D, conv_b, conv_w, dt_bias, in_w, norm_w, out_w
    BM_FWD = 18,
    CV1 = 26, CV2 = 44, CV3 = 62, CV4 = 80,
    CR_EBLN_G = 98, CR_EBLN_B, CR_EINW, CR_ELN_G, CR_ELN_B,
    CR_EM = 103,
    CR_FFN_B1 = 111, CR_FFN_B2, CR_FFN_W1, CR_FFN_W2,
    CR_LN1_G, CR_LN1_B, CR_LN2_G, CR_LN2_B, CR_LN3_G, CR_LN3_B,
    CR_SBLN_G, CR_SBLN_B,
    CR_SM = 123,
    DENSE_B = 131, DENSE_W = 132
};

struct MambaW {
    const float *A_log, *D, *conv_b, *conv_w, *dt_bias, *in_w, *norm_w, *out_w;
    int N, P;
};
MambaW getM(void* const* din, int b, int N, int P) {
    MambaW w;
    w.A_log  = (const float*)din[b + 0]; w.D      = (const float*)din[b + 1];
    w.conv_b = (const float*)din[b + 2]; w.conv_w = (const float*)din[b + 3];
    w.dt_bias= (const float*)din[b + 4]; w.in_w   = (const float*)din[b + 5];
    w.norm_w = (const float*)din[b + 6]; w.out_w  = (const float*)din[b + 7];
    w.N = N; w.P = P;
    return w;
}

struct ConvW { const float *w[7], *b[7], *bn1g, *bn1b, *bn2g, *bn2b; int taps[7]; };
ConvW getC(void* const* din, int base) {
    auto F = [&](int i) { return (const float*)din[base + i]; };
    ConvW c;
    // sorted group: bn1_b,bn1_g,bn2_b,bn2_g,c1_b,c1_w,c2_b,c2_w,c3_b,c3_w,
    //               cf_b,cf_w,cl1_b,cl1_w,cl2_b,cl2_w,cr_b,cr_w
    c.bn1b = F(0);  c.bn1g = F(1);  c.bn2b = F(2);  c.bn2g = F(3);
    c.b[0] = F(4);  c.w[0] = F(5);   // c1
    c.b[1] = F(6);  c.w[1] = F(7);   // c2
    c.b[2] = F(8);  c.w[2] = F(9);   // c3
    c.b[5] = F(10); c.w[5] = F(11);  // cf
    c.b[3] = F(12); c.w[3] = F(13);  // cl1
    c.b[4] = F(14); c.w[4] = F(15);  // cl2
    c.b[6] = F(16); c.w[6] = F(17);  // cr
    int tp[7] = {3, 5, 7, 7, 15, 3, 1};
    for (int i = 0; i < 7; ++i) c.taps[i] = tp[i];
    return c;
}

// Mamba2 mixer: out = resid + out_proj(rmsnorm(gate(scan(conv(in_proj(u))))))
void run_mamba2(hipStream_t s, Bump& ws, const _Float16* u16, float* out, int out_ldc,
                const float* resid, const MambaW& w, int M, int seqlen, int d) {
    const int di = 2 * d, H = di / w.P, N = w.N;
    const int d_in = 2 * di + 2 * N + H;
    const int Bb = M / seqlen;
    size_t mk = ws.off;

    _Float16* inw16  = (_Float16*)ws.take(2ull * d_in * d);
    _Float16* outw16 = (_Float16*)ws.take(2ull * d * di);
    cvt16(s, w.in_w, inw16, (long)d_in * d);
    cvt16(s, w.out_w, outw16, (long)d * di);

    float* zx = (float*)ws.take(4ull * M * d_in);
    gemm(s, u16, inw16, nullptr, nullptr, zx, M, d_in, d, d_in, seqlen, 1, 0, 0);

    float* dt = (float*)ws.take(4ull * M * H);
    float* xb = (float*)ws.take(4ull * M * di);
    float* Bm = (float*)ws.take(4ull * M * N);
    float* Cm = (float*)ws.take(4ull * M * N);
    k_dt<<<grid1((long)M * H), 256, 0, s>>>(zx, w.dt_bias, dt, (long)M * H, d_in, 2 * di + 2 * N, H);
    long tot = (long)M * (di + 2 * N);
    k_dwconv<<<grid1(tot), 256, 0, s>>>(zx, w.conv_w, w.conv_b, xb, Bm, Cm,
                                        tot, seqlen, d_in, di, N, 4);
    float* y = (float*)ws.take(4ull * M * di);
    if (N == 16)
        k_scan<16><<<Bb * H, w.P, 0, s>>>(xb, dt, Bm, Cm, w.A_log, w.D, y, seqlen, H, w.P);
    else
        k_scan<64><<<Bb * H, w.P, 0, s>>>(xb, dt, Bm, Cm, w.A_log, w.D, y, seqlen, H, w.P);

    _Float16* g16 = (_Float16*)ws.take(2ull * M * di);
    k_gate_rms<<<M, 256, 0, s>>>(y, zx, d_in, w.norm_w, g16, di);
    gemm(s, g16, outw16, nullptr, resid, out, M, d, di, out_ldc, seqlen, 1, 0, 0);
    ws.off = mk;
}

void run_convblock(hipStream_t s, Bump& ws, const float* xin, float* out,
                   const ConvW& w, int M, int seqlen, int d) {
    size_t mk = ws.off;
    size_t fs = 4ull * M * d, hs = 2ull * M * d;
    _Float16* in16 = (_Float16*)ws.take(hs);
    float* xa = (float*)ws.take(fs); _Float16* xa16 = (_Float16*)ws.take(hs);
    float* xb = (float*)ws.take(fs); _Float16* xb16 = (_Float16*)ws.take(hs);
    float* xc = (float*)ws.take(fs); _Float16* xc16 = (_Float16*)ws.take(hs);
    float* t1 = (float*)ws.take(fs);
    float* mean = (float*)ws.take(4ull * d);
    float* var  = (float*)ws.take(4ull * d);
    _Float16* w16 = (_Float16*)ws.take(2ull * 15 * d * d);

    auto conv = [&](const _Float16* src16, int ci, const float* resid, float* dst, int act) {
        k_convw<<<grid1((long)d * d * w.taps[ci]), 256, 0, s>>>(w.w[ci], w16, d, d, w.taps[ci]);
        gemm(s, src16, w16, w.b[ci], resid, dst, M, d, d, d,
             seqlen, w.taps[ci], (w.taps[ci] - 1) / 2, act);
    };

    cvt16(s, xin, in16, (long)M * d);
    conv(in16, 0, nullptr, xa, 1);  cvt16(s, xa, xa16, (long)M * d);   // xa = gelu(c1)
    conv(xa16, 1, nullptr, xb, 1);  cvt16(s, xb, xb16, (long)M * d);   // gelu(c2(xa))
    conv(xb16, 2, nullptr, xb, 1);                                     // xb = gelu(c3(..))
    conv(xa16, 3, nullptr, xc, 1);  cvt16(s, xc, xc16, (long)M * d);   // gelu(cl1(xa))
    conv(xc16, 4, xb, xa, 1);       cvt16(s, xa, xa16, (long)M * d);   // xa = gelu(cl2)+xb
    conv(xa16, 5, nullptr, t1, 1);                                     // gelu(cf)
    k_bn_stats<<<d, 256, 0, s>>>(t1, mean, var, M, d);
    k_bn_apply<<<grid1((long)M * d), 256, 0, s>>>(t1, mean, var, w.bn1g, w.bn1b, 0,
                                                  nullptr, xa16, (long)M * d, d);
    conv(xa16, 6, nullptr, t1, 1);                                     // gelu(cr)
    k_bn_stats<<<d, 256, 0, s>>>(t1, mean, var, M, d);
    k_bn_apply<<<grid1((long)M * d), 256, 0, s>>>(t1, mean, var, w.bn2g, w.bn2b, 1,
                                                  out, nullptr, (long)M * d, d);
    ws.off = mk;
}

} // namespace

extern "C" void kernel_launch(void* const* d_in, const int* in_sizes, int n_in,
                              void* d_out, int out_size, void* d_ws, size_t ws_size,
                              hipStream_t stream) {
    (void)in_sizes; (void)n_in; (void)out_size; (void)ws_size;
    const int Bb = 4, L = 2048, d = 256, E = 128;
    const int M = Bb * L;
    const size_t fs = 4ull * M * d, hs = 2ull * M * d;
    hipStream_t s = stream;
    Bump ws{(char*)d_ws, 0};
    auto FI = [&](int i) { return (const float*)d_in[i]; };
    const float* X  = FI(IN_X);
    const float* XE = FI(IN_XE);

    // ---------------- bimamba -> xa0 ----------------
    float* xa0 = (float*)ws.take(fs);
    {
        size_t mk = ws.off;
        float* xn = (float*)ws.take(fs); _Float16* xn16 = (_Float16*)ws.take(hs);
        k_layernorm<<<M, 256, 0, s>>>(X, nullptr, FI(BM_LN1_G), FI(BM_LN1_B), xn, xn16, d);

        float* bi = (float*)ws.take(4ull * M * 2 * d);
        run_mamba2(s, ws, xn16, bi, 2 * d, nullptr, getM(d_in, BM_FWD, 16, 64), M, L, d);

        float* xnr = (float*)ws.take(fs); _Float16* xnr16 = (_Float16*)ws.take(hs);
        k_flip<<<grid1((long)M * d), 256, 0, s>>>(xnr, d, xn, d, (long)M * d, L, d);
        cvt16(s, xnr, xnr16, (long)M * d);
        float* tb = (float*)ws.take(fs);
        run_mamba2(s, ws, xnr16, tb, d, nullptr, getM(d_in, BM_BWD, 16, 64), M, L, d);
        k_flip<<<grid1((long)M * d), 256, 0, s>>>(bi + d, 2 * d, tb, d, (long)M * d, L, d);

        _Float16* bi16 = (_Float16*)ws.take(2ull * M * 2 * d);
        cvt16(s, bi, bi16, (long)M * 2 * d);
        _Float16* fw1 = (_Float16*)ws.take(2ull * (4 * d) * (2 * d));
        cvt16(s, FI(BM_FFN_W1), fw1, (long)4 * d * 2 * d);
        float* hb = (float*)ws.take(4ull * M * 4 * d);
        gemm(s, bi16, fw1, FI(BM_FFN_B1), nullptr, hb, M, 4 * d, 2 * d, 4 * d, L, 1, 0, 1);
        _Float16* hb16 = (_Float16*)ws.take(2ull * M * 4 * d);
        cvt16(s, hb, hb16, (long)M * 4 * d);
        _Float16* fw2 = (_Float16*)ws.take(2ull * d * (4 * d));
        cvt16(s, FI(BM_FFN_W2), fw2, (long)d * 4 * d);
        float* t2 = (float*)ws.take(fs);
        gemm(s, hb16, fw2, FI(BM_FFN_B2), X, t2, M, d, 4 * d, d, L, 1, 0, 1);
        k_layernorm<<<M, 256, 0, s>>>(t2, nullptr, FI(BM_LN2_G), FI(BM_LN2_B), xa0, nullptr, d);
        ws.off = mk;
    }

    // ---------------- conv1 -> xa1; conv2 -> xa_skip ----------------
    float* xa1   = (float*)ws.take(fs);
    float* xskip = (float*)ws.take(fs);
    run_convblock(s, ws, xa0, xa1, getC(d_in, CV1), M, L, d);
    run_convblock(s, ws, xa1, xskip, getC(d_in, CV2), M, L, d);

    // ---------------- dense -> xad; conv3 -> xa3 ----------------
    float* xad = (float*)ws.take(fs);
    {
        size_t mk = ws.off;
        _Float16* a16 = (_Float16*)ws.take(hs);
        cvt16(s, xa1, a16, (long)M * d);
        _Float16* dw = (_Float16*)ws.take(2ull * d * d);
        cvt16(s, FI(DENSE_W), dw, (long)d * d);
        gemm(s, a16, dw, FI(DENSE_B), nullptr, xad, M, d, d, d, L, 1, 0, 1);
        ws.off = mk;
    }
    float* xa3 = (float*)ws.take(fs);
    run_convblock(s, ws, xad, xa3, getC(d_in, CV3), M, L, d);

    // ---------------- cross(local=xa3, glob=xa0, x_extra) -> xcr ----------------
    float* xcr = (float*)ws.take(fs);
    {
        size_t mk = ws.off;
        float* lnl = (float*)ws.take(fs);
        float* lng = (float*)ws.take(fs);
        k_layernorm<<<M, 256, 0, s>>>(xa3, nullptr, FI(CR_LN1_G), FI(CR_LN1_B), lnl, nullptr, d);
        k_layernorm<<<M, 256, 0, s>>>(xa0, nullptr, FI(CR_LN2_G), FI(CR_LN2_B), lng, nullptr, d);

        const int ME = Bb * E;
        float* he  = (float*)ws.take(4ull * ME * d);
        float* he2 = (float*)ws.take(4ull * ME * d);
        k_outer<<<grid1((long)ME * d), 256, 0, s>>>(XE, FI(CR_EINW), he, (long)ME * d, d);
        k_layernorm<<<ME, 256, 0, s>>>(he, nullptr, FI(CR_ELN_G), FI(CR_ELN_B), he2, nullptr, d);
        _Float16* heln16 = (_Float16*)ws.take(2ull * ME * d);
        k_layernorm<<<ME, 256, 0, s>>>(he2, nullptr, FI(CR_EBLN_G), FI(CR_EBLN_B),
                                       nullptr, heln16, d);
        float* extra = (float*)ws.take(4ull * ME * d);
        run_mamba2(s, ws, heln16, extra, d, he2, getM(d_in, CR_EM, 64, 128), ME, E, d);

        const int T2 = E + 2 * L;     // 4224
        const int M2 = Bb * T2;       // 16896
        float* xx = (float*)ws.take(4ull * M2 * d);
        k_copy_seq<<<grid1((long)ME * d), 256, 0, s>>>(xx, T2, 0, extra, E, 0, nullptr,
                                                       (long)ME * d, E, d);
        k_copy_seq<<<grid1((long)M * d), 256, 0, s>>>(xx, T2, E, lng, L, 0, nullptr,
                                                      (long)M * d, L, d);
        k_copy_seq<<<grid1((long)M * d), 256, 0, s>>>(xx, T2, E + L, lnl, L, 0, nullptr,
                                                      (long)M * d, L, d);
        _Float16* xxln16 = (_Float16*)ws.take(2ull * M2 * d);
        k_layernorm<<<M2, 256, 0, s>>>(xx, nullptr, FI(CR_SBLN_G), FI(CR_SBLN_B),
                                       nullptr, xxln16, d);
        float* xx2 = (float*)ws.take(4ull * M2 * d);
        run_mamba2(s, ws, xxln16, xx2, d, xx, getM(d_in, CR_SM, 64, 128), M2, T2, d);

        float* xsl = (float*)ws.take(fs);
        k_copy_seq<<<grid1((long)M * d), 256, 0, s>>>(xsl, L, 0, xx2, T2, E + L, lnl,
                                                      (long)M * d, L, d);
        float* xxn = (float*)ws.take(fs);
        _Float16* xxn16 = (_Float16*)ws.take(hs);
        k_layernorm<<<M, 256, 0, s>>>(xsl, nullptr, FI(CR_LN3_G), FI(CR_LN3_B), xxn, xxn16, d);

        _Float16* w1 = (_Float16*)ws.take(2ull * (d / 2) * d);
        cvt16(s, FI(CR_FFN_W1), w1, (long)(d / 2) * d);
        float* hb = (float*)ws.take(4ull * M * (d / 2));
        gemm(s, xxn16, w1, FI(CR_FFN_B1), nullptr, hb, M, d / 2, d, d / 2, L, 1, 0, 1);
        _Float16* hb16 = (_Float16*)ws.take(2ull * M * (d / 2));
        cvt16(s, hb, hb16, (long)M * (d / 2));
        _Float16* w2 = (_Float16*)ws.take(2ull * d * (d / 2));
        cvt16(s, FI(CR_FFN_W2), w2, (long)d * (d / 2));
        gemm(s, hb16, w2, FI(CR_FFN_B2), xxn, xcr, M, d, d / 2, d, L, 1, 0, 0);
        ws.off = mk;
    }

    // ---------------- conv4; final concat ----------------
    float* xa4 = (float*)ws.take(fs);
    run_convblock(s, ws, xcr, xa4, getC(d_in, CV4), M, L, d);
    k_cat2<<<grid1((long)M * 2 * d), 256, 0, s>>>((float*)d_out, xskip, xa4, (long)M, d);
}
